// MoEProjection_28613072126135
// MI455X (gfx1250) — matrix operand order repaired and checked by reading, not compiled, not run on hardware
//
#include <hip/hip_runtime.h>
#include <hip/hip_bf16.h>

typedef __attribute__((ext_vector_type(16))) _Float16 v16h;
typedef __attribute__((ext_vector_type(8)))  _Float16 v8h;
typedef __attribute__((ext_vector_type(8)))  float    v8f;

#define E_    4
#define NQ_   144
#define NL_   2
#define NH_   16
#define H_    1024
#define OUT_  4096
#define TD_   256
#define B_    8
#define LV_   576
#define LP_   192
#define HD_   64
#define GIN_  (H_ + 2*TD_)   // 1536

// ---------------------------------------------------------------------------
// WMMA GEMM (all-f16 operands, f32 accumulate):
//   Out = epi( A[M,K](f16) * W[N,K]^T(f16) + bias[N] )
// One wave computes a 16(M) x 64(N) tile; block = 4 waves = 64x64 tile.
// The 64x32 f16 weight tile (4KB) is staged once per block into LDS with
// async copies (ASYNCcnt path), double-buffered so the next tile's copies
// overlap the current tile's WMMAs. B fragments are direct 32B LDS vector
// loads (no cvt); A fragments are two 16B global loads (no cvt).
// M % 64 == 0, N % 64 == 0, K % 32 == 0 (guaranteed by all call sites).
// epi: 0 = f32 Out = acc+bias
//      1 = f16 Out16 = gelu(acc+bias)
//      2 = o = resid + ls[n]*(acc+bias); f32 Out = o (in-place ok); f16 Out16 = o
//      3 = f32 Out[(r/sq)*nqtot+qo+(r%sq), n] += gates[(r/sq)*E]*(acc+bias)
// ---------------------------------------------------------------------------
#define BT_TILE (64 * 32)      // halfs per B tile (4KB)

__global__ __launch_bounds__(128)
void gemm_wmma_f16(const _Float16* __restrict__ A, const _Float16* __restrict__ W,
                   float* __restrict__ Out, _Float16* __restrict__ Out16,
                   int M, int N, int K,
                   const float* __restrict__ bias, int epi,
                   const float* __restrict__ scale, const float* __restrict__ resid,
                   int sq, int qo, int nqtot)
{
  __shared__ _Float16 Bt[2][BT_TILE];        // 2 x 4KB double buffer

  const int lane = threadIdx.x & 31;
  const int wave = threadIdx.x >> 5;
  const int n0 = blockIdx.x * 64;
  const int m0 = blockIdx.y * 64 + wave * 16;

  // A fragment addressing (lane holds row m, 16 consecutive K; contiguous k order, same as B)
  const int   ar    = m0 + (lane & 15);
  const int   akoff = (lane >> 4) * 16;      // 0 or 16
  const _Float16* Abase = A + (size_t)ar * K + akoff;
  // B fragment addressing: lane holds col n, 16 consecutive K (from LDS tile)
  const int   bnl   = lane & 15;
  const int   bkoff = (lane >> 4) * 16;      // 0 or 16

  // LDS byte offset of the tile buffers (low 32 bits of flat addr == LDS addr)
  const unsigned ldsbase = (unsigned)(size_t)(void*)&Bt[0][0];

  // Issue one 64(rows n) x 32(k) f16 tile as 256 async b128 copies
  // (2 per thread): chunk c -> row = c>>2, 8-half sub-chunk = c&3.
  auto issue_tile = [&](int step, int sel) {
    const _Float16* gbase = W + (size_t)n0 * K + (size_t)step * 32;
    const unsigned lbase = ldsbase + (unsigned)sel * (BT_TILE * 2u);
#pragma unroll
    for (int i = 0; i < 2; ++i) {
      const int c    = i * 128 + (int)threadIdx.x;
      const int row  = c >> 2;
      const int col8 = (c & 3) << 3;
      const _Float16* gp = gbase + (size_t)row * K + col8;
      const unsigned lo = lbase + (unsigned)c * 16u;
      asm volatile("global_load_async_to_lds_b128 %0, %1, off"
                   :: "v"(lo), "v"(gp) : "memory");
    }
  };

  v8f c0 = {}, c1 = {}, c2 = {}, c3 = {};
  const int nsteps = K >> 5;

  issue_tile(0, 0);
  for (int s = 0; s < nsteps; ++s) {
    // drain own async copies, then block-wide barrier => tile s resident
    asm volatile("s_wait_asynccnt 0x0" ::: "memory");
    __syncthreads();
    if (s + 1 < nsteps) issue_tile(s + 1, (s + 1) & 1);

    const _Float16* pa = Abase + (s << 5);
    __builtin_prefetch(pa + 32, 0, 1);
    const v8h alo = *(const v8h*)pa;
    const v8h ahi = *(const v8h*)(pa + 8);
    const v16h a = __builtin_shufflevector(alo, ahi,
                     0,1,2,3,4,5,6,7,8,9,10,11,12,13,14,15);

    const int sel = s & 1;
    v16h bf;
    bf = *(const v16h*)&Bt[sel][(0 * 16 + bnl) * 32 + bkoff];
    c0 = __builtin_amdgcn_wmma_f32_16x16x32_f16(false, a, false, bf, (short)0, c0, false, false);
    bf = *(const v16h*)&Bt[sel][(1 * 16 + bnl) * 32 + bkoff];
    c1 = __builtin_amdgcn_wmma_f32_16x16x32_f16(false, a, false, bf, (short)0, c1, false, false);
    bf = *(const v16h*)&Bt[sel][(2 * 16 + bnl) * 32 + bkoff];
    c2 = __builtin_amdgcn_wmma_f32_16x16x32_f16(false, a, false, bf, (short)0, c2, false, false);
    bf = *(const v16h*)&Bt[sel][(3 * 16 + bnl) * 32 + bkoff];
    c3 = __builtin_amdgcn_wmma_f32_16x16x32_f16(false, a, false, bf, (short)0, c3, false, false);
  }

  v8f cc[4] = {c0, c1, c2, c3};
  const int rbase = m0 + (lane >> 4) * 8;    // D layout: VGPR i -> M = i (+8 hi lanes)
  const int clo   = lane & 15;
#pragma unroll
  for (int t = 0; t < 4; ++t) {
#pragma unroll
    for (int i = 0; i < 8; ++i) {
      const int r = rbase + i;
      const int c = n0 + t * 16 + clo;
      float v = cc[t][i] + (bias ? bias[c] : 0.0f);
      if (epi == 0) {
        Out[(size_t)r * N + c] = v;
      } else if (epi == 1) {
        Out16[(size_t)r * N + c] =
            (_Float16)(0.5f * v * (1.0f + erff(v * 0.70710678118654752f)));
      } else if (epi == 2) {
        const float o = resid[(size_t)r * N + c] + scale[c] * v;
        Out[(size_t)r * N + c]   = o;
        Out16[(size_t)r * N + c] = (_Float16)o;
      } else {
        const int b    = r / sq;
        const int orow = b * nqtot + qo + (r - b * sq);
        Out[(size_t)orow * N + c] += scale[b * E_] * v;
      }
    }
  }
}

// ---------------------------------------------------------------------------
// f32 -> f16 weight staging
// ---------------------------------------------------------------------------
__global__ void cvt_f32_f16(const float* __restrict__ in, _Float16* __restrict__ out,
                            size_t n) {
  const size_t i = (size_t)blockIdx.x * 256 + threadIdx.x;
  if (i < n) out[i] = (_Float16)in[i];
}

// ---------------------------------------------------------------------------
// LayerNorm over rows of width `width` (1024); f32 in, f16 out (GEMM A input)
// ---------------------------------------------------------------------------
__global__ __launch_bounds__(256)
void ln_rows(const float* __restrict__ in, _Float16* __restrict__ out,
             const float* __restrict__ w, const float* __restrict__ b,
             int width, float eps)
{
  __shared__ float red[256];
  const int r = blockIdx.x, tid = threadIdx.x;
  const float* x = in + (size_t)r * width;
  float s = 0.f;
  for (int j = tid; j < width; j += 256) s += x[j];
  red[tid] = s; __syncthreads();
  for (int o = 128; o > 0; o >>= 1) { if (tid < o) red[tid] += red[tid + o]; __syncthreads(); }
  const float mean = red[0] / width; __syncthreads();
  float s2 = 0.f;
  for (int j = tid; j < width; j += 256) { float d = x[j] - mean; s2 += d * d; }
  red[tid] = s2; __syncthreads();
  for (int o = 128; o > 0; o >>= 1) { if (tid < o) red[tid] += red[tid + o]; __syncthreads(); }
  const float rstd = rsqrtf(red[0] / width + eps);
  for (int j = tid; j < width; j += 256)
    out[(size_t)r * width + j] = (_Float16)((x[j] - mean) * rstd * w[j] + b[j]);
}

// ---------------------------------------------------------------------------
// Build stage buffers: x (broadcast query tile) and kv0 = [x_init ; z-slice]
// ---------------------------------------------------------------------------
__global__ __launch_bounds__(256)
void build_stage(const float* __restrict__ query_e, const float* __restrict__ img,
                 const float* __restrict__ pp, float* __restrict__ x,
                 float* __restrict__ kv0, int sq, int Lkv, int qo, int io)
{
  const int b = blockIdx.x / Lkv;
  const int j = blockIdx.x % Lkv;
  const float* src;
  float* dx = nullptr;
  if (j < sq) {
    src = query_e + (size_t)(qo + j) * H_;
    dx = x + ((size_t)b * sq + j) * H_;
  } else {
    const int g = io + (j - sq);
    src = (g < LV_) ? img + ((size_t)b * LV_ + g) * H_
                    : pp  + ((size_t)b * LP_ + (g - LV_)) * H_;
  }
  float* dk = kv0 + ((size_t)b * Lkv + j) * H_;
  for (int t = threadIdx.x; t < H_; t += 256) {
    const float v = src[t];
    dk[t] = v;
    if (dx) dx[t] = v;
  }
}

// ---------------------------------------------------------------------------
// Fused attention: block = (b, head), thread = one query row, flash-style
// qp rows: (b*sq+q)*H + h*64 + d ; kvp rows: (b*Lkv+k)*2H, [K | V]
// Output written as f16 (consumed only as GEMM A input).
// ---------------------------------------------------------------------------
__global__ __launch_bounds__(64)
void attn_fused(const float* __restrict__ qp, const float* __restrict__ kvp,
                _Float16* __restrict__ o, int sqv, int Lkv)
{
  __shared__ float Ks[64 * HD_];
  __shared__ float Vs[64 * HD_];
  const int b = blockIdx.x / NH_;
  const int h = blockIdx.x % NH_;
  const int t = threadIdx.x;

  float q[HD_], oa[HD_];
  float mx = -1e30f, lsum = 0.f;
  if (t < sqv) {
    const float* qr = qp + ((size_t)b * sqv + t) * H_ + h * HD_;
#pragma unroll
    for (int d = 0; d < HD_; ++d) { q[d] = qr[d]; oa[d] = 0.f; }
  }
  for (int c0 = 0; c0 < Lkv; c0 += 64) {
    const int krow = c0 + t;
    if (krow < Lkv) {
      const float* kr = kvp + ((size_t)b * Lkv + krow) * (2 * H_) + h * HD_;
      const float* vr = kr + H_;
      for (int d = 0; d < HD_; ++d) { Ks[t * HD_ + d] = kr[d]; Vs[t * HD_ + d] = vr[d]; }
    }
    __syncthreads();
    const int kmax = (Lkv - c0 < 64) ? (Lkv - c0) : 64;
    if (t < sqv) {
      for (int kk = 0; kk < kmax; ++kk) {
        float s = 0.f;
#pragma unroll
        for (int d = 0; d < HD_; ++d) s += q[d] * Ks[kk * HD_ + d];
        s *= 0.125f;                              // 1/sqrt(64)
        const float m2   = fmaxf(mx, s);
        const float corr = __expf(mx - m2);
        const float p    = __expf(s - m2);
        lsum = lsum * corr + p;
#pragma unroll
        for (int d = 0; d < HD_; ++d) oa[d] = oa[d] * corr + p * Vs[kk * HD_ + d];
        mx = m2;
      }
    }
    __syncthreads();
  }
  if (t < sqv) {
    const float inv = 1.0f / lsum;
    _Float16* orow = o + ((size_t)b * sqv + t) * H_ + h * HD_;
#pragma unroll
    for (int d = 0; d < HD_; ++d) orow[d] = (_Float16)(oa[d] * inv);
  }
}

// ---------------------------------------------------------------------------
// Gating
// ---------------------------------------------------------------------------
__global__ __launch_bounds__(256)
void gate_prep(const float* __restrict__ img, const float* __restrict__ temb,
               const float* __restrict__ eemb, const int* __restrict__ tid,
               const int* __restrict__ eid, float* __restrict__ gi)
{
  const int b = blockIdx.x;
  for (int hh = threadIdx.x; hh < H_; hh += 256) {
    float s = 0.f;
    for (int k = 0; k < LV_; ++k) s += img[((size_t)b * LV_ + k) * H_ + hh];
    gi[(size_t)b * GIN_ + hh] = s * (1.0f / LV_);
  }
  const int tk = tid[b], ek = eid[b];
  for (int j = threadIdx.x; j < TD_; j += 256) {
    gi[(size_t)b * GIN_ + H_ + j]       = temb[tk * TD_ + j];
    gi[(size_t)b * GIN_ + H_ + TD_ + j] = eemb[ek * TD_ + j];
  }
}

__global__ __launch_bounds__(256)
void gate_compute(const float* __restrict__ gi, const float* __restrict__ lnw,
                  const float* __restrict__ lnb, const float* __restrict__ gw,
                  const float* __restrict__ gb, float* __restrict__ gates)
{
  __shared__ float red[256];
  __shared__ float logits[E_];
  const int b = blockIdx.x, tid = threadIdx.x;
  const float* xr = gi + (size_t)b * GIN_;
  float s = 0.f;
  for (int j = tid; j < GIN_; j += 256) s += xr[j];
  red[tid] = s; __syncthreads();
  for (int o = 128; o > 0; o >>= 1) { if (tid < o) red[tid] += red[tid + o]; __syncthreads(); }
  const float mean = red[0] * (1.0f / GIN_); __syncthreads();
  float s2 = 0.f;
  for (int j = tid; j < GIN_; j += 256) { float d = xr[j] - mean; s2 += d * d; }
  red[tid] = s2; __syncthreads();
  for (int o = 128; o > 0; o >>= 1) { if (tid < o) red[tid] += red[tid + o]; __syncthreads(); }
  const float rstd = rsqrtf(red[0] * (1.0f / GIN_) + 1e-5f); __syncthreads();

  for (int e = 0; e < E_; ++e) {
    float p = 0.f;
    for (int j = tid; j < GIN_; j += 256)
      p += ((xr[j] - mean) * rstd * lnw[j] + lnb[j]) * gw[e * GIN_ + j];
    red[tid] = p; __syncthreads();
    for (int o = 128; o > 0; o >>= 1) { if (tid < o) red[tid] += red[tid + o]; __syncthreads(); }
    if (tid == 0) {
      float lg = red[0] + gb[e];
      logits[e] = fminf(15.f, fmaxf(-15.f, lg));
    }
    __syncthreads();
  }
  if (tid == 0) {
    float mxl = logits[0];
    for (int e = 1; e < E_; ++e) mxl = fmaxf(mxl, logits[e]);
    float w[E_]; float sm = 0.f;
    for (int e = 0; e < E_; ++e) { w[e] = __expf(logits[e] - mxl); sm += w[e]; }
    for (int e = 0; e < E_; ++e) w[e] /= sm;
    int i1 = 0;
    for (int e = 1; e < E_; ++e) if (w[e] > w[i1]) i1 = e;        // first-index ties
    int i2 = -1;
    for (int e = 0; e < E_; ++e) if (e != i1 && (i2 < 0 || w[e] > w[i2])) i2 = e;
    const float denom = w[i1] + w[i2] + 1e-9f;
    for (int e = 0; e < E_; ++e) gates[b * E_ + e] = 0.f;
    gates[b * E_ + i1] = w[i1] / denom;
    gates[b * E_ + i2] = w[i2] / denom;
  }
}

// ---------------------------------------------------------------------------
__global__ void zero_f32(float* __restrict__ p, size_t n) {
  const size_t i = (size_t)blockIdx.x * 256 + threadIdx.x;
  if (i < n) p[i] = 0.f;
}

__global__ __launch_bounds__(256)
void final_rms(const float* __restrict__ comb, const float* __restrict__ fw,
               const float* __restrict__ og, float* __restrict__ out)
{
  __shared__ float red[256];
  const size_t r = blockIdx.x;
  const int tid = threadIdx.x;
  const float* x = comb + r * OUT_;
  float s = 0.f;
  for (int j = tid; j < OUT_; j += 256) s += x[j] * x[j];
  red[tid] = s; __syncthreads();
  for (int o = 128; o > 0; o >>= 1) { if (tid < o) red[tid] += red[tid + o]; __syncthreads(); }
  const float rms = rsqrtf(red[0] * (1.0f / OUT_) + 1e-6f);
  for (int j = tid; j < OUT_; j += 256)
    out[r * OUT_ + j] = x[j] * rms * fw[j] * og[j];
}

// ---------------------------------------------------------------------------
extern "C" void kernel_launch(void* const* d_in, const int* in_sizes, int n_in,
                              void* d_out, int out_size, void* d_ws, size_t ws_size,
                              hipStream_t stream)
{
  (void)in_sizes; (void)n_in; (void)out_size; (void)ws_size;
  const float* img        = (const float*)d_in[0];
  const float* pp         = (const float*)d_in[1];
  const int*   task_ids   = (const int*)d_in[2];
  const int*   elem_ids   = (const int*)d_in[3];
  const float* query      = (const float*)d_in[4];
  const float* ln1_w      = (const float*)d_in[5];
  const float* ln1_b      = (const float*)d_in[6];
  const float* ln1kv_w    = (const float*)d_in[7];
  const float* ln1kv_b    = (const float*)d_in[8];
  const float* attn_in_w  = (const float*)d_in[9];
  const float* attn_in_b  = (const float*)d_in[10];
  const float* attn_out_w = (const float*)d_in[11];
  const float* attn_out_b = (const float*)d_in[12];
  const float* ls1        = (const float*)d_in[13];
  const float* ls2        = (const float*)d_in[14];
  const float* ln2_w      = (const float*)d_in[15];
  const float* ln2_b      = (const float*)d_in[16];
  const float* fc_w       = (const float*)d_in[17];
  const float* fc_b       = (const float*)d_in[18];
  const float* proj_w     = (const float*)d_in[19];
  const float* proj_b     = (const float*)d_in[20];
  const float* outp_w     = (const float*)d_in[21];
  const float* outp_b     = (const float*)d_in[22];
  const float* temb       = (const float*)d_in[23];
  const float* eemb       = (const float*)d_in[24];
  const float* gate_ln_w  = (const float*)d_in[25];
  const float* gate_ln_b  = (const float*)d_in[26];
  const float* gate_w     = (const float*)d_in[27];
  const float* gate_b     = (const float*)d_in[28];
  const float* output_gain= (const float*)d_in[29];
  const float* final_w    = (const float*)d_in[30];

  char* ws = (char*)d_ws;
  size_t off = 0;
  auto allocb = [&](size_t bytes) {
    void* p = ws + off;
    off += (bytes + 255) & ~(size_t)255;
    return p;
  };
  float*     gi      = (float*)    allocb((size_t)B_ * GIN_ * 4);
  float*     gates   = (float*)    allocb((size_t)B_ * E_ * 4);
  float*     x       = (float*)    allocb((size_t)B_ * 64 * H_ * 4);
  _Float16*  xn16    = (_Float16*) allocb((size_t)B_ * 64 * H_ * 2);
  float*     kv0     = (float*)    allocb((size_t)B_ * 320 * H_ * 4);
  _Float16*  kvn16   = (_Float16*) allocb((size_t)B_ * 320 * H_ * 2);
  float*     qp      = (float*)    allocb((size_t)B_ * 64 * H_ * 4);
  float*     kvp     = (float*)    allocb((size_t)B_ * 320 * 2 * H_ * 4);
  _Float16*  attno16 = (_Float16*) allocb((size_t)B_ * 64 * H_ * 2);
  _Float16*  h2_16   = (_Float16*) allocb((size_t)B_ * 64 * 4 * H_ * 2);
  _Float16*  xsh16   = (_Float16*) allocb((size_t)B_ * 64 * H_ * 2);
  float*     comb    = (float*)    allocb((size_t)B_ * NQ_ * OUT_ * 4);
  // per-expert f16 weight staging (reused across experts)
  const size_t nAttnIn  = (size_t)NL_ * 3 * H_ * H_;
  const size_t nAttnOut = (size_t)NL_ * H_ * H_;
  const size_t nFc      = (size_t)NL_ * 4 * H_ * H_;
  const size_t nProj    = (size_t)NL_ * 4 * H_ * H_;
  const size_t nOutp    = (size_t)OUT_ * H_;
  _Float16* wAttnIn  = (_Float16*)allocb(nAttnIn  * 2);
  _Float16* wAttnOut = (_Float16*)allocb(nAttnOut * 2);
  _Float16* wFc      = (_Float16*)allocb(nFc      * 2);
  _Float16* wProj    = (_Float16*)allocb(nProj    * 2);
  _Float16* wOutp    = (_Float16*)allocb(nOutp    * 2);

  // gating + clear accumulator (must be re-zeroed every call: deterministic)
  gate_prep<<<B_, 256, 0, stream>>>(img, temb, eemb, task_ids, elem_ids, gi);
  gate_compute<<<B_, 256, 0, stream>>>(gi, gate_ln_w, gate_ln_b, gate_w, gate_b, gates);
  const size_t ncomb = (size_t)B_ * NQ_ * OUT_;
  zero_f32<<<(unsigned)((ncomb + 255) / 256), 256, 0, stream>>>(comb, ncomb);

  const int stage_q[3]  = {64, 48, 32};
  const int stage_io[3] = {0, 256, 512};

  for (int e = 0; e < E_; ++e) {
    // stage this expert's weights as f16
    cvt_f32_f16<<<(unsigned)((nAttnIn + 255) / 256), 256, 0, stream>>>(
        attn_in_w + (size_t)e * nAttnIn, wAttnIn, nAttnIn);
    cvt_f32_f16<<<(unsigned)((nAttnOut + 255) / 256), 256, 0, stream>>>(
        attn_out_w + (size_t)e * nAttnOut, wAttnOut, nAttnOut);
    cvt_f32_f16<<<(unsigned)((nFc + 255) / 256), 256, 0, stream>>>(
        fc_w + (size_t)e * nFc, wFc, nFc);
    cvt_f32_f16<<<(unsigned)((nProj + 255) / 256), 256, 0, stream>>>(
        proj_w + (size_t)e * nProj, wProj, nProj);
    cvt_f32_f16<<<(unsigned)((nOutp + 255) / 256), 256, 0, stream>>>(
        outp_w + (size_t)e * nOutp, wOutp, nOutp);

    int qo = 0;
    for (int s = 0; s < 3; ++s) {
      const int sq  = stage_q[s];
      const int io  = stage_io[s];
      const int Lkv = sq + 256;
      const int Mq  = B_ * sq;    // 512/384/256  (all %64==0)
      const int Mkv = B_ * Lkv;   // 2560/2432/2304

      build_stage<<<B_ * Lkv, 256, 0, stream>>>(query + (size_t)e * NQ_ * H_,
                                                img, pp, x, kv0, sq, Lkv, qo, io);
      for (int l = 0; l < NL_; ++l) {
        const size_t vH  = (size_t)(e * NL_ + l) * H_;
        const _Float16* WinQ  = wAttnIn + (size_t)l * 3 * H_ * H_;
        const float*    bin   = attn_in_b + (size_t)(e * NL_ + l) * 3 * H_;

        ln_rows<<<Mkv, 256, 0, stream>>>(kv0, kvn16, ln1kv_w + vH, ln1kv_b + vH, H_, 1e-5f);
        ln_rows<<<Mq,  256, 0, stream>>>(x,   xn16,  ln1_w + vH,   ln1_b + vH,   H_, 1e-5f);

        // Q projection
        gemm_wmma_f16<<<dim3(H_ / 64, Mq / 64), 128, 0, stream>>>(
            xn16, WinQ, qp, nullptr, Mq, H_, H_, bin, 0, nullptr, nullptr, 0, 0, 0);
        // K and V projection (rows H..3H are contiguous)
        gemm_wmma_f16<<<dim3((2 * H_) / 64, Mkv / 64), 128, 0, stream>>>(
            kvn16, WinQ + (size_t)H_ * H_, kvp, nullptr, Mkv, 2 * H_, H_, bin + H_, 0,
            nullptr, nullptr, 0, 0, 0);

        attn_fused<<<B_ * NH_, 64, 0, stream>>>(qp, kvp, attno16, sq, Lkv);

        // out-proj with fused residual + layerscale 1 (in-place on x, f16 shadow)
        gemm_wmma_f16<<<dim3(H_ / 64, Mq / 64), 128, 0, stream>>>(
            attno16, wAttnOut + (size_t)l * H_ * H_, x, xsh16, Mq, H_, H_,
            attn_out_b + vH, 2, ls1 + vH, x, 0, 0, 0);

        ln_rows<<<Mq, 256, 0, stream>>>(x, xn16, ln2_w + vH, ln2_b + vH, H_, 1e-5f);

        // MLP up with fused exact GELU -> f16 h2
        gemm_wmma_f16<<<dim3((4 * H_) / 64, Mq / 64), 128, 0, stream>>>(
            xn16, wFc + (size_t)l * 4 * H_ * H_, nullptr, h2_16, Mq, 4 * H_, H_,
            fc_b + (size_t)(e * NL_ + l) * 4 * H_, 1, nullptr, nullptr, 0, 0, 0);
        // MLP down with fused residual + layerscale 2 (in-place on x, f16 shadow)
        gemm_wmma_f16<<<dim3(H_ / 64, Mq / 64), 128, 0, stream>>>(
            h2_16, wProj + (size_t)l * H_ * 4 * H_, x, xsh16, Mq, H_, 4 * H_,
            proj_b + vH, 2, ls2 + vH, x, 0, 0, 0);
      }
      // expert output projection, gate-weighted accumulate into comb
      // (xsh16 is the f16 shadow of x written by the last epi-2 above)
      gemm_wmma_f16<<<dim3(OUT_ / 64, Mq / 64), 128, 0, stream>>>(
          xsh16, wOutp, comb, nullptr, Mq, OUT_, H_,
          outp_b + (size_t)e * OUT_, 3, gates + e, nullptr, sq, qo, NQ_);
      qo += sq;
    }
  }

  final_rms<<<B_ * NQ_, 256, 0, stream>>>(comb, final_w, output_gain, (float*)d_out);
}
